// GFLSTM_80504866996851
// MI455X (gfx1250) — matrix operand order repaired and checked by reading, not compiled, not run on hardware
//
#include <hip/hip_runtime.h>
#include <hip/hip_bf16.h>

// Problem constants (match reference)
#define T_  256
#define B_  32
#define D_  512
#define H_  512
#define L_  3
#define H3_ 1536   // 3*H
#define LH_ 1536   // L*H

#define NWG      32
#define NTHREADS 256   // 8 wave32 per workgroup -> 256 waves total

// stage-1 task segmentation (per layer):
//   [0,384)   : gg half-K tiles   (32 WMMA each)
//   [384,576) : gate tiles        (32 WMMA each)
//   [576,768) : uc tiles          (16 WMMA each)
//   [768,832) : wc tiles          (16 WMMA each)
#define NTASKS_ 832

typedef __attribute__((ext_vector_type(16))) __bf16 v16bf;
typedef __attribute__((ext_vector_type(8)))  __bf16 v8bf;
typedef __attribute__((ext_vector_type(8)))  float  v8f;

__device__ __forceinline__ float sigmoidf_(float x) { return 1.0f / (1.0f + __expf(-x)); }

// One K-strip GEMM accumulation pass on a 16x16 tile with v_wmma_f32_16x16x32_bf16.
// A: row-major [M x K] activations (ldA), Bt: transposed weights [N x K] (ldB, K contiguous).
__device__ __forceinline__ v8f wmma_pass(v8f acc,
                                         const __bf16* __restrict__ A, int ldA,
                                         const __bf16* __restrict__ Bt, int ldB,
                                         int mbase, int nbase, int K, int lane)
{
    const int  l15 = lane & 15;
    const bool hi  = (lane >= 16);
    // A 16x32 bf16 fragment layout (ISA 7.12.2): lanes 0-15 hold K 0-7 & 16-23 of row M=lane,
    // lanes 16-31 hold K 8-15 & 24-31 of row M=lane-16.
    const __bf16* ap = A  + (size_t)(mbase + l15) * ldA + (hi ? 8 : 0);
    // B 32x16 fragment: same K order as A (lanes 0-15: K 0-7 & 16-23, lanes 16-31: K 8-15 & 24-31).
    const __bf16* bp = Bt + (size_t)(nbase + l15) * ldB + (hi ? 8 : 0);

    for (int k0 = 0; k0 < K; k0 += 32) {
        __builtin_prefetch(bp + k0 + 512, 0, 3);   // global_prefetch_b8, keep-near
        v8bf a0 = *(const v8bf*)(ap + k0);
        v8bf a1 = *(const v8bf*)(ap + k0 + 16);
        v16bf a;
#pragma unroll
        for (int i = 0; i < 8; ++i) { a[i] = a0[i]; a[i + 8] = a1[i]; }
        v8bf b0 = *(const v8bf*)(bp + k0);
        v8bf b1 = *(const v8bf*)(bp + k0 + 16);
        v16bf b;
#pragma unroll
        for (int i = 0; i < 8; ++i) { b[i] = b0[i]; b[i + 8] = b1[i]; }
        acc = __builtin_amdgcn_wmma_f32_16x16x32_bf16(
            /*neg_a=*/false, a, /*neg_b=*/false, b,
            /*c_mod=*/(short)0, acc, /*reuse_a=*/false, /*reuse_b=*/false);
    }
    return acc;
}

// Device-scope barrier across the persistent grid (monotonic generation counter).
__device__ __forceinline__ void gbar(unsigned* cnt, unsigned* gen)
{
    __syncthreads();
    if (threadIdx.x == 0) {
        __threadfence();
        atomicAdd(cnt, 1u);
        const unsigned target = (*gen + 1u) * (unsigned)gridDim.x;
        volatile unsigned* vc = (volatile unsigned*)cnt;
        while (*vc < target) __builtin_amdgcn_s_sleep(1);
        __threadfence();
    }
    ++(*gen);
    __syncthreads();
}

struct Params {
    // fp32 biases (read directly from inputs)
    const float *b_i2h, *b_h2h, *b_wc, *b_wg, *b_ug, *b_uc;
    // bf16 converted inputs / transposed weights (in workspace)
    const __bf16 *xbf;                       // [T][B][D]
    const __bf16 *Wi2hT, *Wh2hT, *WwcT;      // [L][N][K]
    const __bf16 *WwgT, *WugT, *WucT;        // [L][L][N][K]
    __bf16 *hbf;                             // [2][L][B][H] ping-pong hidden (bf16)
    float  *cbuf;                            // [L][B][H] cell state (fp32, in-place)
    float  *hbuf;                            // [L][B][H] last hidden (fp32, for h_n)
    float  *gates;                           // [B][3H]   post-sigmoid
    float  *preWC;                           // [B][H]    raw inp@W_wc
    float  *preGGa;                          // [L][B][H] gg partial (inp + h0 K-half)
    float  *preGGb;                          // [L][B][H] gg partial (h1 + h2 K-half)
    float  *preUC;                           // [L][B][H] raw h_lp@W_uc
    unsigned *barrier;
    float  *out;                             // [T*B*H | L*B*H | L*B*H]
};

__global__ void gflstm_persistent(Params p)
{
    const int lane   = threadIdx.x & 31;
    const int waveId = (int)((blockIdx.x * blockDim.x + threadIdx.x) >> 5);
    const int nWaves = (int)((gridDim.x * blockDim.x) >> 5);
    const int tid    = (int)(blockIdx.x * blockDim.x + threadIdx.x);
    const int nThr   = (int)(gridDim.x * blockDim.x);
    const v8f vzero  = {0.f, 0.f, 0.f, 0.f, 0.f, 0.f, 0.f, 0.f};
    unsigned gen = 0;

    for (int t = 0; t < T_; ++t) {
        const int cur = t & 1, nxt = cur ^ 1;
        const __bf16* hprev = p.hbf + (size_t)cur * (L_ * B_ * H_);
        __bf16*       hnext = p.hbf + (size_t)nxt * (L_ * B_ * H_);

        for (int l = 0; l < L_; ++l) {
            const __bf16* inp = (l == 0) ? (p.xbf + (size_t)t * B_ * D_)
                                         : (hnext + (size_t)(l - 1) * B_ * H_);

            // ---------------- stage 1: uniform WMMA tile tasks ----------------
            for (int task = waveId; task < NTASKS_; task += nWaves) {
                if (task < 384) {
                    // gg partial: half of the K=2048 accumulation for sigmoid-gate gg[lp]
                    const int half = task & 1;
                    const int tmp  = task >> 1;            // lp*64 + tile
                    const int lp   = tmp >> 6;
                    const int tile = tmp & 63;
                    const int nbase = (tile >> 1) * 16, mbase = (tile & 1) * 16;
                    v8f acc = vzero;
                    if (half == 0) {
                        // inp @ W_wg[l][lp] (K=512)  +  h0 @ W_ug[l][lp][0:512] (K=512)
                        acc = wmma_pass(acc, inp, D_,
                                        p.WwgT + (size_t)(l * L_ + lp) * H_ * D_, D_,
                                        mbase, nbase, D_, lane);
                        acc = wmma_pass(acc, hprev, H_,
                                        p.WugT + (size_t)(l * L_ + lp) * H_ * LH_, LH_,
                                        mbase, nbase, H_, lane);
                    } else {
                        // h1 @ W_ug[...][512:1024]  +  h2 @ W_ug[...][1024:1536]
                        acc = wmma_pass(acc, hprev + (size_t)1 * B_ * H_, H_,
                                        p.WugT + (size_t)(l * L_ + lp) * H_ * LH_ + 1 * H_, LH_,
                                        mbase, nbase, H_, lane);
                        acc = wmma_pass(acc, hprev + (size_t)2 * B_ * H_, H_,
                                        p.WugT + (size_t)(l * L_ + lp) * H_ * LH_ + 2 * H_, LH_,
                                        mbase, nbase, H_, lane);
                    }
                    const int n    = nbase + (lane & 15);
                    const int mofs = mbase + ((lane >= 16) ? 8 : 0);
                    float* dst = (half == 0 ? p.preGGa : p.preGGb) + (size_t)lp * B_ * H_;
#pragma unroll
                    for (int r = 0; r < 8; ++r)
                        dst[(size_t)(mofs + r) * H_ + n] = acc[r];
                } else if (task < 576) {
                    // gates = sigmoid([inp|h_l] @ [W_i2h;W_h2h] + b): K fused 1024
                    const int t2 = task - 384;
                    const int nbase = (t2 >> 1) * 16, mbase = (t2 & 1) * 16;
                    v8f acc = vzero;
                    acc = wmma_pass(acc, inp, D_,
                                    p.Wi2hT + (size_t)l * H3_ * D_, D_, mbase, nbase, D_, lane);
                    acc = wmma_pass(acc, hprev + (size_t)l * B_ * H_, H_,
                                    p.Wh2hT + (size_t)l * H3_ * H_, H_, mbase, nbase, H_, lane);
                    const int n    = nbase + (lane & 15);
                    const int mofs = mbase + ((lane >= 16) ? 8 : 0);
                    const float bb = p.b_i2h[l * H3_ + n] + p.b_h2h[l * H3_ + n];
#pragma unroll
                    for (int r = 0; r < 8; ++r)
                        p.gates[(size_t)(mofs + r) * H3_ + n] = sigmoidf_(acc[r] + bb);
                } else if (task < 768) {
                    // uc raw: h_lp @ W_uc[l][lp] (K=512)
                    const int t3   = task - 576;
                    const int lp   = t3 >> 6;
                    const int tile = t3 & 63;
                    const int nbase = (tile >> 1) * 16, mbase = (tile & 1) * 16;
                    v8f acc = vzero;
                    acc = wmma_pass(acc, hprev + (size_t)lp * B_ * H_, H_,
                                    p.WucT + (size_t)(l * L_ + lp) * H_ * H_, H_,
                                    mbase, nbase, H_, lane);
                    const int n    = nbase + (lane & 15);
                    const int mofs = mbase + ((lane >= 16) ? 8 : 0);
#pragma unroll
                    for (int r = 0; r < 8; ++r)
                        p.preUC[((size_t)lp * B_ + mofs + r) * H_ + n] = acc[r];
                } else {
                    // wc raw: inp @ W_wc[l] (K=512)
                    const int tile = task - 768;
                    const int nbase = (tile >> 1) * 16, mbase = (tile & 1) * 16;
                    v8f acc = vzero;
                    acc = wmma_pass(acc, inp, D_,
                                    p.WwcT + (size_t)l * H_ * D_, D_, mbase, nbase, D_, lane);
                    const int n    = nbase + (lane & 15);
                    const int mofs = mbase + ((lane >= 16) ? 8 : 0);
#pragma unroll
                    for (int r = 0; r < 8; ++r)
                        p.preWC[(size_t)(mofs + r) * H_ + n] = acc[r];
                }
            }
            gbar(p.barrier, &gen);

            // ---------------- stage 2: gated-feedback reduction + LSTM cell ----------------
            for (int e = tid; e < B_ * H_; e += nThr) {
                const int b = e >> 9, n = e & (H_ - 1);
                float s = p.preWC[(size_t)b * H_ + n] + p.b_wc[l * H_ + n];
#pragma unroll
                for (int lp = 0; lp < L_; ++lp) {
                    const size_t gi = ((size_t)lp * B_ + b) * H_ + n;
                    const float g = sigmoidf_(p.preGGa[gi] + p.preGGb[gi]
                                              + p.b_wg[(l * L_ + lp) * H_ + n]
                                              + p.b_ug[(l * L_ + lp) * H_ + n]);
                    s += g * (p.preUC[gi] + p.b_uc[(l * L_ + lp) * H_ + n]);
                }
                const float cd = tanhf(s);
                const float ig = p.gates[(size_t)b * H3_ + n];
                const float fg = p.gates[(size_t)b * H3_ + H_ + n];
                const float og = p.gates[(size_t)b * H3_ + 2 * H_ + n];
                const size_t ci = (size_t)(l * B_ + b) * H_ + n;
                const float cn = fg * p.cbuf[ci] + ig * cd;
                const float hn = og * tanhf(cn);
                p.cbuf[ci] = cn;
                p.hbuf[ci] = hn;
                hnext[ci]  = (__bf16)hn;
                if (l == L_ - 1) p.out[(size_t)t * B_ * H_ + e] = hn;
            }
            gbar(p.barrier, &gen);
        }
    }

    // final states: h_n then c_n appended after the [T,B,H] sequence output
    for (int e = tid; e < L_ * B_ * H_; e += nThr) {
        p.out[(size_t)T_ * B_ * H_ + e]                = p.hbuf[e];
        p.out[(size_t)T_ * B_ * H_ + L_ * B_ * H_ + e] = p.cbuf[e];
    }
}

// ---- one-time prep: fp32 -> bf16 conversions ----
__global__ void prep_convert(const float* __restrict__ x, const float* __restrict__ h0,
                             const float* __restrict__ c0,
                             __bf16* __restrict__ xbf, __bf16* __restrict__ hbf0,
                             float* __restrict__ cbuf)
{
    const size_t tid  = (size_t)blockIdx.x * blockDim.x + threadIdx.x;
    const size_t nThr = (size_t)gridDim.x * blockDim.x;
    for (size_t i = tid; i < (size_t)T_ * B_ * D_; i += nThr) xbf[i] = (__bf16)x[i];
    for (size_t i = tid; i < (size_t)L_ * B_ * H_; i += nThr) {
        hbf0[i] = (__bf16)h0[i];
        cbuf[i] = c0[i];
    }
}

// convert + transpose [nmat][K][N] fp32 -> [nmat][N][K] bf16
__global__ void prep_transpose(const float* __restrict__ src, __bf16* __restrict__ dst,
                               int nmat, int K, int N)
{
    const size_t total = (size_t)nmat * K * N;
    const size_t tid   = (size_t)blockIdx.x * blockDim.x + threadIdx.x;
    const size_t nThr  = (size_t)gridDim.x * blockDim.x;
    for (size_t i = tid; i < total; i += nThr) {
        const size_t m = i / ((size_t)K * N);
        const size_t r = i - m * (size_t)K * N;
        const size_t k = r / (size_t)N, n = r - k * (size_t)N;
        dst[m * (size_t)K * N + n * (size_t)K + k] = (__bf16)src[i];
    }
}

extern "C" void kernel_launch(void* const* d_in, const int* in_sizes, int n_in,
                              void* d_out, int out_size, void* d_ws, size_t ws_size,
                              hipStream_t stream)
{
    (void)in_sizes; (void)n_in; (void)out_size; (void)ws_size;

    const float* x     = (const float*)d_in[0];
    const float* h0    = (const float*)d_in[1];
    const float* c0    = (const float*)d_in[2];
    const float* W_i2h = (const float*)d_in[3];
    const float* b_i2h = (const float*)d_in[4];
    const float* W_h2h = (const float*)d_in[5];
    const float* b_h2h = (const float*)d_in[6];
    const float* W_wc  = (const float*)d_in[7];
    const float* b_wc  = (const float*)d_in[8];
    const float* W_wg  = (const float*)d_in[9];
    const float* b_wg  = (const float*)d_in[10];
    const float* W_ug  = (const float*)d_in[11];
    const float* b_ug  = (const float*)d_in[12];
    const float* W_uc  = (const float*)d_in[13];
    const float* b_uc  = (const float*)d_in[14];

    char* ws = (char*)d_ws;
    size_t off = 0;
    auto alloc = [&](size_t bytes) -> void* {
        off = (off + 255) & ~(size_t)255;
        void* p = ws + off;
        off += bytes;
        return p;
    };

    __bf16* xbf    = (__bf16*)alloc((size_t)T_ * B_ * D_ * 2);
    __bf16* Wi2hT  = (__bf16*)alloc((size_t)L_ * H3_ * D_ * 2);
    __bf16* Wh2hT  = (__bf16*)alloc((size_t)L_ * H3_ * H_ * 2);
    __bf16* WwcT   = (__bf16*)alloc((size_t)L_ * H_ * D_ * 2);
    __bf16* WwgT   = (__bf16*)alloc((size_t)L_ * L_ * H_ * D_ * 2);
    __bf16* WugT   = (__bf16*)alloc((size_t)L_ * L_ * H_ * LH_ * 2);
    __bf16* WucT   = (__bf16*)alloc((size_t)L_ * L_ * H_ * H_ * 2);
    __bf16* hbf    = (__bf16*)alloc((size_t)2 * L_ * B_ * H_ * 2);
    float*  cbuf   = (float*) alloc((size_t)L_ * B_ * H_ * 4);
    float*  hbuf   = (float*) alloc((size_t)L_ * B_ * H_ * 4);
    float*  gates  = (float*) alloc((size_t)B_ * H3_ * 4);
    float*  preWC  = (float*) alloc((size_t)B_ * H_ * 4);
    float*  preGGa = (float*) alloc((size_t)L_ * B_ * H_ * 4);
    float*  preGGb = (float*) alloc((size_t)L_ * B_ * H_ * 4);
    float*  preUC  = (float*) alloc((size_t)L_ * B_ * H_ * 4);
    unsigned* barr = (unsigned*)alloc(256);

    hipMemsetAsync(barr, 0, 256, stream);

    prep_convert<<<1024, 256, 0, stream>>>(x, h0, c0, xbf, hbf, cbuf);
    prep_transpose<<<2048, 256, 0, stream>>>(W_i2h, Wi2hT, L_,      D_,  H3_);
    prep_transpose<<<2048, 256, 0, stream>>>(W_h2h, Wh2hT, L_,      H_,  H3_);
    prep_transpose<<<1024, 256, 0, stream>>>(W_wc,  WwcT,  L_,      D_,  H_);
    prep_transpose<<<2048, 256, 0, stream>>>(W_wg,  WwgT,  L_ * L_, D_,  H_);
    prep_transpose<<<4096, 256, 0, stream>>>(W_ug,  WugT,  L_ * L_, LH_, H_);
    prep_transpose<<<2048, 256, 0, stream>>>(W_uc,  WucT,  L_ * L_, H_,  H_);

    Params P;
    P.b_i2h = b_i2h; P.b_h2h = b_h2h; P.b_wc = b_wc;
    P.b_wg  = b_wg;  P.b_ug  = b_ug;  P.b_uc = b_uc;
    P.xbf = xbf;
    P.Wi2hT = Wi2hT; P.Wh2hT = Wh2hT; P.WwcT = WwcT;
    P.WwgT  = WwgT;  P.WugT  = WugT;  P.WucT = WucT;
    P.hbf = hbf; P.cbuf = cbuf; P.hbuf = hbuf;
    P.gates = gates; P.preWC = preWC;
    P.preGGa = preGGa; P.preGGb = preGGb; P.preUC = preUC;
    P.barrier = barr;
    P.out = (float*)d_out;

    gflstm_persistent<<<NWG, NTHREADS, 0, stream>>>(P);
}
